// MultiHeadSelfAttention_420906795249
// MI455X (gfx1250) — matrix operand order repaired and checked by reading, not compiled, not run on hardware
//
#include <hip/hip_runtime.h>
#include <hip/hip_bf16.h>

// MHSA forward, B=4 N=2048 D=768 H=12 Hd=64.
// bf16 WMMA (v_wmma_f32_16x16x32_bf16), flash-style streaming softmax,
// TDM (tensor_load_to_lds) double-buffered LDS staging of tiles shared by
// all 4 waves of a workgroup, register-only P^T via xor-16 shuffles.

#define EMBED 768
#define NSEQ  2048
#define NBATCH 4
#define NHEAD 12
#define HDIM  64
#define MROWS (NBATCH * NSEQ) // 8192

typedef __attribute__((ext_vector_type(16))) __bf16    v16bf;
typedef __attribute__((ext_vector_type(8)))  float     v8f;
typedef __attribute__((ext_vector_type(4)))  unsigned  v4u;
typedef __attribute__((ext_vector_type(8)))  unsigned  v8u;
typedef __attribute__((ext_vector_type(4)))  int       v4i;
typedef __attribute__((ext_vector_type(8)))  int       v8i;

#if defined(__has_builtin)
#if __has_builtin(__builtin_amdgcn_tensor_load_to_lds)
#define USE_TDM 1
#endif
#endif
#ifndef USE_TDM
#define USE_TDM 0
#endif

__device__ __forceinline__ unsigned short f2bf_bits(float f) {
    unsigned u = __builtin_bit_cast(unsigned, f);
    unsigned r = u + 0x7FFFu + ((u >> 16) & 1u);   // round-to-nearest-even
    return (unsigned short)(r >> 16);
}
__device__ __forceinline__ unsigned pack2(float lo, float hi) {
    return (unsigned)f2bf_bits(lo) | ((unsigned)f2bf_bits(hi) << 16);
}
__device__ __forceinline__ v8f vzero8() {
    v8f z;
#pragma unroll
    for (int i = 0; i < 8; ++i) z[i] = 0.0f;
    return z;
}
__device__ __forceinline__ v16bf make_frag(v4u lo, v4u hi) {
    v8u r;
    r[0] = lo[0]; r[1] = lo[1]; r[2] = lo[2]; r[3] = lo[3];
    r[4] = hi[0]; r[5] = hi[1]; r[6] = hi[2]; r[7] = hi[3];
    return __builtin_bit_cast(v16bf, r);
}

// A-matrix fragment 16x32 bf16 (rows M=0..15, K=0..31), base row-major,
// stride in elements (multiple of 8). Lanes 0-15: K 0-15; lanes 16-31: K 16-31.
__device__ __forceinline__ v16bf load_afrag(const unsigned short* base, int stride, int lane) {
    const int m  = lane & 15;
    const int h8 = (lane >> 4) << 4;       // 0 or 16
    const unsigned short* row = base + (size_t)m * stride;
    v4u lo = *(const v4u*)(row + h8);
    v4u hi = *(const v4u*)(row + 8 + h8);
    return make_frag(lo, hi);
}

// B-matrix fragment 32x16 bf16 (K=0..31, N=0..15). Element (k,n) at
// base + n*stride + k (column-contiguous over k). Lanes 0-15: K 0-15; 16-31: K 16-31.
__device__ __forceinline__ v16bf load_bfrag(const unsigned short* base, int stride, int lane) {
    const int n   = lane & 15;
    const int k16 = (lane >> 4) << 4;      // 0 or 16
    const unsigned short* col = base + (size_t)n * stride + k16;
    v4u lo = *(const v4u*)(col);
    v4u hi = *(const v4u*)(col + 8);
    return make_frag(lo, hi);
}

__device__ __forceinline__ v8f wmma_bf16(v16bf a, v16bf b, v8f c) {
    return __builtin_amdgcn_wmma_f32_16x16x32_bf16(
        false, a, false, b, (short)0, c, false, false);
}

// ---------------------------------------------------------------------------
// TDM: async 2D-tile load Global -> LDS (D# per CDNA5 ISA).
// tile_x contiguous elements (2B each) per row, tile_y rows, row stride
// stride_x elements. Issued by one wave; completion via s_wait_tensorcnt.
// ---------------------------------------------------------------------------
#if USE_TDM
__device__ __forceinline__ void tdm_load_2d(unsigned lds_byte_off, const void* gaddr,
                                            unsigned tile_x, unsigned tile_y,
                                            unsigned tensor_x, unsigned tensor_y,
                                            unsigned stride_x) {
    unsigned long long ga = (unsigned long long)gaddr;
    v4u g0;
    g0[0] = 1u;                                        // count=1, user descriptor
    g0[1] = lds_byte_off;                              // lds_addr
    g0[2] = (unsigned)(ga & 0xFFFFFFFFu);              // global_addr[31:0]
    g0[3] = (unsigned)((ga >> 32) & 0x01FFFFFFu) | (2u << 30);  // addr[56:32] | type=2
    v8i g1;
    g1[0] = (int)(1u << 16);                           // data_size=1 -> 2 bytes
    g1[1] = (int)((tensor_x & 0xFFFFu) << 16);         // tensor_dim0[15:0]
    g1[2] = (int)((tensor_x >> 16) | ((tensor_y & 0xFFFFu) << 16));
    g1[3] = (int)((tensor_y >> 16) | (tile_x << 16));  // tile_dim0
    g1[4] = (int)tile_y;                               // tile_dim1 (tile_dim2=0)
    g1[5] = (int)stride_x;                             // tensor_dim0_stride[31:0]
    g1[6] = 0;
    g1[7] = 0;
    v4i z4 = {0, 0, 0, 0};
#if __clang_major__ >= 23
    v8i z8 = {0, 0, 0, 0, 0, 0, 0, 0};
    __builtin_amdgcn_tensor_load_to_lds(g0, g1, z4, z4, z8, 0);
#else
    __builtin_amdgcn_tensor_load_to_lds(g0, g1, z4, z4, 0);
#endif
}
__device__ __forceinline__ void tdm_wait() { __builtin_amdgcn_s_wait_tensorcnt(0); }
#else
__device__ __forceinline__ void tdm_wait() {}
#endif

// Fallback: cooperative global->LDS tile copy (all threads of the block).
__device__ __forceinline__ void coop_copy_tile(unsigned short* dst, const unsigned short* src,
                                               int rows, int xlen, int gstride) {
    const int chunks = rows * xlen / 8;
    for (int c = threadIdx.x; c < chunks; c += blockDim.x) {
        const int r = (c * 8) / xlen;
        const int x = (c * 8) % xlen;
        *(v4u*)(dst + r * xlen + x) = *(const v4u*)(src + (size_t)r * gstride + x);
    }
}

// ---------------------------------------------------------------------------
// fp32 -> bf16 conversion (pairs, n even)
// ---------------------------------------------------------------------------
__global__ void cvt_f32_bf16(const float* __restrict__ src,
                             unsigned short* __restrict__ dst, int n) {
    int idx = (blockIdx.x * blockDim.x + threadIdx.x) * 2;
    if (idx + 1 < n) *(unsigned*)(dst + idx) = pack2(src[idx], src[idx + 1]);
}

// ---------------------------------------------------------------------------
// GEMM: Out[M,768] = A[M,768](bf16) @ W[768,768]^T(bf16) + bias.
// WG = 128 rows x 64 cols (4 waves x 32x64, 8 accumulators each).
// W tile (64 cols x 32 k) staged in LDS via TDM, double-buffered.
// MODE 0: Q (b,h,n,hd) bf16 *1/8 | 1: K (b,h,n,hd) | 2: V^T (b,h,hd,n) | 3: f32 row-major
// ---------------------------------------------------------------------------
template <int MODE>
__global__ __launch_bounds__(128) void gemm_kernel(
    const unsigned short* __restrict__ A, const unsigned short* __restrict__ W,
    const float* __restrict__ bias, unsigned short* __restrict__ outb,
    float* __restrict__ outf) {
    __shared__ unsigned short wsm[2 * 64 * 32];        // [buf][col][k], 8KB
    const int lane = threadIdx.x & 31;
    const int wave = __builtin_amdgcn_readfirstlane(threadIdx.x >> 5);  // wave-uniform
    const int row0 = blockIdx.x * 128 + wave * 32;
    const int col0 = blockIdx.y * 64;
    const unsigned short* Wt = W + (size_t)col0 * EMBED;

    // stage first W tile
#if USE_TDM
    if (wave == 0) tdm_load_2d(0u, Wt, 32, 64, EMBED, EMBED, EMBED);
    tdm_wait();
#else
    coop_copy_tile(wsm, Wt, 64, 32, EMBED);
#endif
    __syncthreads();

    v8f acc[2][4];
#pragma unroll
    for (int r = 0; r < 2; ++r)
#pragma unroll
        for (int j = 0; j < 4; ++j) acc[r][j] = vzero8();

#pragma unroll 2
    for (int k0 = 0; k0 < EMBED; k0 += 32) {
        const int cur = (k0 >> 5) & 1;
        const bool more = (k0 + 32) < EMBED;
        if (more) {                                    // stage next W tile
#if USE_TDM
            if (wave == 0)
                tdm_load_2d((unsigned)((cur ^ 1) * 4096), Wt + k0 + 32, 32, 64,
                            EMBED, EMBED, EMBED);
#else
            coop_copy_tile(wsm + (cur ^ 1) * 2048, Wt + k0 + 32, 64, 32, EMBED);
#endif
            __builtin_prefetch(A + (size_t)row0 * EMBED + k0 + 32 + (size_t)lane * EMBED, 0, 1);
        }
        v16bf a0 = load_afrag(A + (size_t)row0 * EMBED + k0, EMBED, lane);
        v16bf a1 = load_afrag(A + (size_t)(row0 + 16) * EMBED + k0, EMBED, lane);
#pragma unroll
        for (int j = 0; j < 4; ++j) {
            v16bf b = load_bfrag(wsm + cur * 2048 + 16 * j * 32, 32, lane);
            acc[0][j] = wmma_bf16(a0, b, acc[0][j]);
            acc[1][j] = wmma_bf16(a1, b, acc[1][j]);
        }
        if (more) {
            tdm_wait();
            __syncthreads();
        }
    }

    const int rl = (lane >> 4) << 3;   // 0 or 8
    const int cl = lane & 15;
#pragma unroll
    for (int rr = 0; rr < 2; ++rr)
#pragma unroll
        for (int j = 0; j < 4; ++j)
#pragma unroll
            for (int i = 0; i < 8; ++i) {
                const int r = row0 + 16 * rr + i + rl;
                const int c = col0 + 16 * j + cl;
                float v = acc[rr][j][i] + bias[c];
                if (MODE == 0) v *= 0.125f;            // fold 1/sqrt(Hd) into Q
                if (MODE == 3) {
                    outf[(size_t)r * EMBED + c] = v;
                } else {
                    const int bb = r >> 11, n = r & (NSEQ - 1);
                    const int h = c >> 6, hd = c & (HDIM - 1);
                    size_t idx;
                    if (MODE == 2)
                        idx = ((size_t)(bb * NHEAD + h) * HDIM + hd) * NSEQ + n;
                    else
                        idx = ((size_t)(bb * NHEAD + h) * NSEQ + n) * HDIM + hd;
                    outb[idx] = f2bf_bits(v);
                }
            }
}

// ---------------------------------------------------------------------------
// Flash attention. WG = 64 queries (4 waves x 16q x Hd=64); key blocks of 32.
// K (32x64) and V^T (64x32) tiles shared by all waves: TDM -> LDS, dbl-buffered.
// S^T = K(A) x Q^T(B);  O^T = V^T(A) x P^T(B), P^T built via xor-16 shuffles.
// ---------------------------------------------------------------------------
__global__ __launch_bounds__(128) void attn_kernel(
    const unsigned short* __restrict__ Q, const unsigned short* __restrict__ K,
    const unsigned short* __restrict__ Vt, unsigned short* __restrict__ Ob) {
    __shared__ unsigned short smem[2 * 32 * 64 + 2 * 64 * 32];   // K bufs + V bufs, 16KB
    unsigned short* kbuf = smem;                 // [buf][key][hd]  stride 64
    unsigned short* vbuf = smem + 4096;          // [buf][hd][key]  stride 32

    const int lane = threadIdx.x & 31;
    const int wave = __builtin_amdgcn_readfirstlane(threadIdx.x >> 5);  // wave-uniform
    const int bh = blockIdx.y;
    const int q0 = blockIdx.x * 64 + wave * 16;

    const unsigned short* Qb = Q + (size_t)bh * NSEQ * HDIM;
    const unsigned short* Kb = K + (size_t)bh * NSEQ * HDIM;
    const unsigned short* Vb = Vt + (size_t)bh * HDIM * NSEQ;

    // Q^T B-fragments (hd 0..31 and 32..63), reused across all key blocks
    const v16bf qf0 = load_bfrag(Qb + (size_t)q0 * HDIM, HDIM, lane);
    const v16bf qf1 = load_bfrag(Qb + (size_t)q0 * HDIM + 32, HDIM, lane);

    // stage first K/V^T tiles
#if USE_TDM
    if (wave == 0) {
        tdm_load_2d(0u, Kb, 64, 32, 64, NSEQ, 64);                 // K block 0
        tdm_load_2d(8192u, Vb, 32, 64, NSEQ, 64, NSEQ);            // V^T block 0
    }
    tdm_wait();
#else
    coop_copy_tile(kbuf, Kb, 32, 64, 64);
    coop_copy_tile(vbuf, Vb, 64, 32, NSEQ);
#endif
    __syncthreads();

    v8f o[4];
#pragma unroll
    for (int j = 0; j < 4; ++j) o[j] = vzero8();
    float m = -1e30f, l = 0.0f;
    const v8f zc = vzero8();
    const bool hihalf = lane >= 16;

#pragma unroll 2
    for (int k0 = 0; k0 < NSEQ; k0 += 32) {
        const int cur = (k0 >> 5) & 1;
        const bool more = (k0 + 32) < NSEQ;
        if (more) {                                    // stage next tiles
#if USE_TDM
            if (wave == 0) {
                tdm_load_2d((unsigned)((cur ^ 1) * 4096),
                            Kb + (size_t)(k0 + 32) * HDIM, 64, 32, 64, NSEQ, 64);
                tdm_load_2d((unsigned)(8192 + (cur ^ 1) * 4096),
                            Vb + k0 + 32, 32, 64, NSEQ, 64, NSEQ);
            }
#else
            coop_copy_tile(kbuf + (cur ^ 1) * 2048, Kb + (size_t)(k0 + 32) * HDIM, 32, 64, 64);
            coop_copy_tile(vbuf + (cur ^ 1) * 2048, Vb + k0 + 32, 64, 32, NSEQ);
#endif
        }
        const unsigned short* kt = kbuf + cur * 2048;
        const unsigned short* vt = vbuf + cur * 2048;

        v16bf ka00 = load_afrag(kt, 64, lane);                 // keys 0-15, hd 0-31
        v16bf ka01 = load_afrag(kt + 32, 64, lane);            // keys 0-15, hd 32-63
        v16bf ka10 = load_afrag(kt + 16 * 64, 64, lane);       // keys 16-31, hd 0-31
        v16bf ka11 = load_afrag(kt + 16 * 64 + 32, 64, lane);

        // S^T tiles: rows keys, cols queries (Q already carries 1/sqrt(Hd))
        v8f s0 = wmma_bf16(ka01, qf1, wmma_bf16(ka00, qf0, zc));
        v8f s1 = wmma_bf16(ka11, qf1, wmma_bf16(ka10, qf0, zc));

        // online softmax per query column (lane + its xor-16 partner)
        float vmax = s0[0];
#pragma unroll
        for (int i = 1; i < 8; ++i) vmax = fmaxf(vmax, s0[i]);
#pragma unroll
        for (int i = 0; i < 8; ++i) vmax = fmaxf(vmax, s1[i]);
        vmax = fmaxf(vmax, __shfl_xor(vmax, 16, 32));
        const float mnew = fmaxf(m, vmax);
        const float alpha = __expf(m - mnew);

        float p0[8], p1[8], rs = 0.0f;
#pragma unroll
        for (int i = 0; i < 8; ++i) { p0[i] = __expf(s0[i] - mnew); rs += p0[i]; }
#pragma unroll
        for (int i = 0; i < 8; ++i) { p1[i] = __expf(s1[i] - mnew); rs += p1[i]; }
        rs += __shfl_xor(rs, 16, 32);
        l = l * alpha + rs;
        m = mnew;
#pragma unroll
        for (int j = 0; j < 4; ++j)
#pragma unroll
            for (int i = 0; i < 8; ++i) o[j][i] *= alpha;

        // Build P^T (32 keys x 16 queries) B-fragment in registers.
        unsigned pk0[4], pk1[4];
#pragma unroll
        for (int t = 0; t < 4; ++t) {
            pk0[t] = pack2(p0[2 * t], p0[2 * t + 1]);
            pk1[t] = pack2(p1[2 * t], p1[2 * t + 1]);
        }
        v8u pr;
#pragma unroll
        for (int t = 0; t < 4; ++t) {
            unsigned sh0 = (unsigned)__shfl_xor((int)pk0[t], 16, 32);
            unsigned sh1 = (unsigned)__shfl_xor((int)pk1[t], 16, 32);
            pr[t]     = hihalf ? sh1 : pk0[t];   // keys 0-7 | 16-23
            pr[4 + t] = hihalf ? pk1[t] : sh0;   // keys 8-15 | 24-31
        }
        v16bf pb = __builtin_bit_cast(v16bf, pr);

        // O^T += V^T x P^T  (4 hd blocks of 16)
#pragma unroll
        for (int j = 0; j < 4; ++j) {
            v16bf va = load_afrag(vt + 16 * j * 32, 32, lane);
            o[j] = wmma_bf16(va, pb, o[j]);
        }
        if (more) {
            tdm_wait();
            __syncthreads();
        }
    }

    const float inv = 1.0f / l;
    const int b = bh / NHEAD, h = bh % NHEAD;
    const int qcol = lane & 15;
    const int hoff = (lane >> 4) << 3;
    const size_t rowg = (size_t)(b * NSEQ + q0 + qcol) * EMBED;
#pragma unroll
    for (int j = 0; j < 4; ++j)
#pragma unroll
        for (int t = 0; t < 4; ++t) {
            const int hd = h * HDIM + 16 * j + hoff + 2 * t;
            *(unsigned*)(Ob + rowg + hd) = pack2(o[j][2 * t] * inv, o[j][2 * t + 1] * inv);
        }
}

// ---------------------------------------------------------------------------
extern "C" void kernel_launch(void* const* d_in, const int* in_sizes, int n_in,
                              void* d_out, int out_size, void* d_ws, size_t ws_size,
                              hipStream_t stream) {
    const float* x     = (const float*)d_in[0];
    const float* W_q   = (const float*)d_in[1];
    const float* b_q   = (const float*)d_in[2];
    const float* W_k   = (const float*)d_in[3];
    const float* b_k   = (const float*)d_in[4];
    const float* W_v   = (const float*)d_in[5];
    const float* b_v   = (const float*)d_in[6];
    const float* W_out = (const float*)d_in[7];
    const float* b_out = (const float*)d_in[8];
    float* out = (float*)d_out;

    const size_t nX = (size_t)MROWS * EMBED;        // 6291456
    const size_t nW = (size_t)EMBED * EMBED;        // 589824
    unsigned short* xb  = (unsigned short*)d_ws;
    unsigned short* wqb = xb + nX;
    unsigned short* wkb = wqb + nW;
    unsigned short* wvb = wkb + nW;
    unsigned short* wob = wvb + nW;
    unsigned short* Qw  = wob + nW;
    unsigned short* Kw  = Qw + nX;
    unsigned short* Vtw = Kw + nX;
    unsigned short* Obw = Vtw + nX;

    // 1) convert inputs to bf16
    cvt_f32_bf16<<<(int)((nX / 2 + 255) / 256), 256, 0, stream>>>(x, xb, (int)nX);
    cvt_f32_bf16<<<(int)((nW / 2 + 255) / 256), 256, 0, stream>>>(W_q, wqb, (int)nW);
    cvt_f32_bf16<<<(int)((nW / 2 + 255) / 256), 256, 0, stream>>>(W_k, wkb, (int)nW);
    cvt_f32_bf16<<<(int)((nW / 2 + 255) / 256), 256, 0, stream>>>(W_v, wvb, (int)nW);
    cvt_f32_bf16<<<(int)((nW / 2 + 255) / 256), 256, 0, stream>>>(W_out, wob, (int)nW);

    // 2) QKV projections (WMMA bf16, TDM-staged W tiles)
    dim3 ggrid(MROWS / 128, EMBED / 64);
    gemm_kernel<0><<<ggrid, 128, 0, stream>>>(xb, wqb, b_q, Qw, nullptr);
    gemm_kernel<1><<<ggrid, 128, 0, stream>>>(xb, wkb, b_k, Kw, nullptr);
    gemm_kernel<2><<<ggrid, 128, 0, stream>>>(xb, wvb, b_v, Vtw, nullptr);

    // 3) flash attention (WMMA bf16, TDM-staged K/V^T tiles)
    attn_kernel<<<dim3(NSEQ / 64, NBATCH * NHEAD), 128, 0, stream>>>(Qw, Kw, Vtw, Obw);

    // 4) output projection -> fp32
    gemm_kernel<3><<<ggrid, 128, 0, stream>>>(Obw, wob, b_out, nullptr, out);
}
